// REM_5609227289004
// MI455X (gfx1250) — matrix operand order repaired and checked by reading, not compiled, not run on hardware
//
#include <hip/hip_runtime.h>

// MI455X / gfx1250 wave32 WMMA pipeline:
//   conv1 -> conv2 -> conv3 -> lin1 -> 200 heads, all implicit-GEMM via
//   v_wmma_f32_16x16x32_f16 (f16 operands, f32 accumulate).
// LDS tiles are stored in *fragment order* so operand loads are ds_load_b128:
//   A (16-bit, 16x32): lane's 16 halves live at row*LDH + half*16 .. +16,
//   where element k is stored at position swapbits3_4(k) within the row.
//   B (32x16): stored [n][k] per column, same swapbits3_4(k) permutation as A.
// Staging uses global_load_async_to_lds_b128 (ASYNCcnt path; builtin confirmed
// present on this toolchain) with plain vector-copy fallback.

typedef __attribute__((ext_vector_type(16))) _Float16 v16h;
typedef __attribute__((ext_vector_type(8)))  _Float16 v8h;
typedef __attribute__((ext_vector_type(8)))  float    v8f;

typedef int v4i_vs __attribute__((vector_size(16)));          // int __vector(4)
typedef __attribute__((address_space(1))) v4i_vs* gv4i_p;     // global (AS1)
typedef __attribute__((address_space(3))) v4i_vs* lv4i_p;     // LDS (AS3)

#if defined(__has_builtin)
#if __has_builtin(__builtin_amdgcn_global_load_async_to_lds_b128) && \
    __has_builtin(__builtin_amdgcn_s_wait_asynccnt)
#define USE_ASYNC_LDS 1
#endif
#endif
#ifndef USE_ASYNC_LDS
#define USE_ASYNC_LDS 0
#endif

// move 16 bytes global -> LDS (per-lane)
__device__ __forceinline__ void stage16B(const _Float16* g, _Float16* l) {
#if USE_ASYNC_LDS
    __builtin_amdgcn_global_load_async_to_lds_b128((gv4i_p)g, (lv4i_p)l, 0, 0);
#else
    *(v8h*)l = *(const v8h*)g;
#endif
}
__device__ __forceinline__ void stage_wait() {
#if USE_ASYNC_LDS
    __builtin_amdgcn_s_wait_asynccnt(0);
#endif
}

// one contiguous 32B fragment row -> v16h (two ds_load_b128)
__device__ __forceinline__ v16h load_frag(const _Float16* p) {
    const v8h* vp = (const v8h*)p;
    v8h a = vp[0], b = vp[1];
    return __builtin_shufflevector(a, b, 0, 1, 2, 3, 4, 5, 6, 7,
                                   8, 9, 10, 11, 12, 13, 14, 15);
}

// ---------------------------------------------------------------- weight cvt
__global__ __launch_bounds__(256)
void cvt_f32_to_f16(const float* __restrict__ in, _Float16* __restrict__ out, int n) {
    int i = blockIdx.x * 256 + threadIdx.x;
    if (i < n) out[i] = (_Float16)in[i];
}

// heads_w [200,512,18] f32 -> transposed [200,18,512] f16 (contiguous-K for GEMM B)
__global__ __launch_bounds__(256)
void cvt_heads_T(const float* __restrict__ in, _Float16* __restrict__ out) {
    int i = blockIdx.x * 256 + threadIdx.x;     // over 200*512*18
    if (i < 200 * 512 * 18) {
        int head = i / 9216;
        int rem  = i - head * 9216;
        int d    = rem / 18;
        int a    = rem - d * 18;
        out[head * 9216 + a * 512 + d] = (_Float16)in[i];
    }
}

// ------------------------------------------------------- implicit-GEMM conv
// out[m,n] = relu( sum_k im2col(in)[m,k] * w[n,k] + bias[n] ), stored NCHW f16.
// VEC_A: A is plain row-major f16 [M,Ktot] (lin1 path) -> vector staging.
template<int NTILE, typename TIn, bool VEC_A>
__global__ __launch_bounds__(128)
void conv_gemm_wmma(const TIn* __restrict__ in, const _Float16* __restrict__ w,
                    const float* __restrict__ bias, _Float16* __restrict__ out,
                    int Ktot, int Cout,
                    int HWout, int Wout, int strideS, int KHW, int KW,
                    int inCHW, int inHW, int Win)
{
    constexpr int NF  = NTILE / 16;
    constexpr int LDH = 40;                      // halves/row: 80B, 16B aligned
    __shared__ _Float16 As[64 * LDH];
    __shared__ _Float16 Bs[NTILE * LDH];

    const int t     = threadIdx.x;
    const int wv    = t >> 5;                    // wave -> rows [wv*16, wv*16+16)
    const int lane  = t & 31;
    const int lhi   = lane >> 4;
    const int l16   = lane & 15;
    const int mBase = blockIdx.x * 64;
    const int nBase = blockIdx.y * NTILE;

    // A staging geometry: kk fixed per thread (128 % 32 == 0), rows rowA+4j
    const int kkA  = t & 31;
    const int rowA = t >> 5;
    const int posA = (kkA & 7) | ((kkA >> 1) & 8) | ((kkA << 1) & 16); // swap b3<->b4

    // per-thread im2col pixel offsets (k-independent), computed once
    int offA[16];
    if constexpr (!VEC_A) {
        int m   = mBase + rowA;
        int b   = m / HWout;
        int pix = m - b * HWout;
        int oh  = pix / Wout;
        int ow  = pix - oh * Wout;
        #pragma unroll
        for (int j = 0; j < 16; ++j) {
            offA[j] = b * inCHW + (oh * strideS) * Win + ow * strideS;
            ow += 4;                                  // 4 < Wout for all conv layers
            if (ow >= Wout) { ow -= Wout; oh += 1; }
            pix += 4;
            if (pix >= HWout) { pix -= HWout; b += 1; oh = 0; ow = pix; }
        }
    }

    v8f acc[NF];
    #pragma unroll
    for (int f = 0; f < NF; ++f)
        #pragma unroll
        for (int e = 0; e < 8; ++e) acc[f][e] = 0.0f;

    for (int k0 = 0; k0 < Ktot; k0 += 32) {
        // ---- stage A (64 x 32 halves, fragment-permuted) ----
        if constexpr (VEC_A) {
            #pragma unroll
            for (int vi = 0; vi < 2; ++vi) {
                int idx = t + vi * 128;
                int row = idx >> 2, g = idx & 3;
                int gp  = ((g & 1) << 1) | (g >> 1);   // group perm = swap b3<->b4
                stage16B(&in[(size_t)(mBase + row) * Ktot + k0 + g * 8],
                         &As[row * LDH + gp * 8]);
            }
        } else {
            int k  = k0 + kkA;
            int c  = k / KHW;
            int r  = k - c * KHW;
            int kh = r / KW;
            int kw = r - kh * KW;
            const int srcOff = c * inHW + kh * Win + kw;
            #pragma unroll
            for (int j = 0; j < 16; ++j)
                As[(rowA + 4 * j) * LDH + posA] = (_Float16)in[srcOff + offA[j]];
        }
        // ---- stage B (NTILE cols x 32 halves, fragment-permuted) ----
        #pragma unroll
        for (int vi = 0; vi < NTILE / 32; ++vi) {
            int idx = t + vi * 128;
            int n = idx >> 2, g = idx & 3;
            const _Float16* src = &w[(size_t)(nBase + n) * Ktot + k0 + g * 8];
            stage16B(src, &Bs[n * LDH + (((g & 1) << 1) | (g >> 1)) * 8]);
#if !USE_ASYNC_LDS
            if (k0 + 32 < Ktot) __builtin_prefetch(src + 32, 0, 1);
#endif
        }
        stage_wait();
        __syncthreads();

        // ---- fragments + WMMA ----
        v16h af = load_frag(&As[(wv * 16 + l16) * LDH + lhi * 16]);
        #pragma unroll
        for (int nf = 0; nf < NF; ++nf) {
            v16h bf = load_frag(&Bs[(nf * 16 + l16) * LDH + lhi * 16]);
            acc[nf] = __builtin_amdgcn_wmma_f32_16x16x32_f16(
                false, af, false, bf, (short)0, acc[nf], false, false);
        }
        __syncthreads();
    }

    // ---- epilogue: bias + relu, NCHW f16 ----
    #pragma unroll
    for (int nf = 0; nf < NF; ++nf) {
        int   n  = nBase + nf * 16 + l16;
        float bv = bias[n];
        #pragma unroll
        for (int r = 0; r < 8; ++r) {
            int   m   = mBase + wv * 16 + lhi * 8 + r;
            float val = acc[nf][r] + bv;
            val = val > 0.0f ? val : 0.0f;
            int b   = m / HWout;
            int pix = m - b * HWout;
            out[(size_t)b * Cout * HWout + (size_t)n * HWout + pix] = (_Float16)val;
        }
    }
}

// ------------------------------------------------------------- 200 heads
// out[(head*1024+b)*18 + a] = h[b,:].heads_w[head,:,a] + heads_b[head,a]
// whT is pre-transposed [200,18,512]; N padded 18->32 with zeroed B rows.
__global__ __launch_bounds__(128)
void heads_wmma(const _Float16* __restrict__ h, const _Float16* __restrict__ whT,
                const float* __restrict__ bh, float* __restrict__ out)
{
    constexpr int LDH = 40;
    __shared__ _Float16 As[64 * LDH];
    __shared__ _Float16 Bs[32 * LDH];

    const int t = threadIdx.x;
    const int wv = t >> 5, lane = t & 31, lhi = lane >> 4, l16 = lane & 15;
    const int mBase = blockIdx.x * 64;
    const int head  = blockIdx.y;
    const _Float16* wk = whT + (size_t)head * 18 * 512;

    {   // zero Bs once (rows 18..31 stay zero through the K loop)
        v8h zv;
        #pragma unroll
        for (int e = 0; e < 8; ++e) zv[e] = (_Float16)0.0f;
        for (int i = t; i < (32 * LDH) / 8; i += 128) ((v8h*)Bs)[i] = zv;
    }
    __syncthreads();

    v8f acc0, acc1;
    #pragma unroll
    for (int e = 0; e < 8; ++e) { acc0[e] = 0.0f; acc1[e] = 0.0f; }

    for (int k0 = 0; k0 < 512; k0 += 32) {
        #pragma unroll
        for (int vi = 0; vi < 2; ++vi) {          // A: 64 rows x 4 groups
            int idx = t + vi * 128;
            int row = idx >> 2, g = idx & 3;
            int gp  = ((g & 1) << 1) | (g >> 1);
            stage16B(&h[(size_t)(mBase + row) * 512 + k0 + g * 8],
                     &As[row * LDH + gp * 8]);
        }
        if (t < 72) {                             // B: 18 rows x 4 groups
            int n = t >> 2, g = t & 3;
            stage16B(&wk[(size_t)n * 512 + k0 + g * 8], &Bs[n * LDH + (((g & 1) << 1) | (g >> 1)) * 8]);
        }
        stage_wait();
        __syncthreads();

        v16h af  = load_frag(&As[(wv * 16 + l16) * LDH + lhi * 16]);
        v16h bf0 = load_frag(&Bs[l16 * LDH + lhi * 16]);
        v16h bf1 = load_frag(&Bs[(16 + l16) * LDH + lhi * 16]);
        acc0 = __builtin_amdgcn_wmma_f32_16x16x32_f16(false, af, false, bf0, (short)0, acc0, false, false);
        acc1 = __builtin_amdgcn_wmma_f32_16x16x32_f16(false, af, false, bf1, (short)0, acc1, false, false);
        __syncthreads();
    }

    float b0 = bh[head * 18 + l16];
    #pragma unroll
    for (int r = 0; r < 8; ++r) {
        int m = mBase + wv * 16 + lhi * 8 + r;
        size_t base = ((size_t)head * 1024 + m) * 18;
        out[base + l16] = acc0[r] + b0;
        if (l16 < 2)
            out[base + 16 + l16] = acc1[r] + bh[head * 18 + 16 + l16];
    }
}

// ---------------------------------------------------------------- launcher
extern "C" void kernel_launch(void* const* d_in, const int* in_sizes, int n_in,
                              void* d_out, int out_size, void* d_ws, size_t ws_size,
                              hipStream_t stream) {
    const float* x       = (const float*)d_in[0];
    const float* conv1_w = (const float*)d_in[1];
    const float* conv1_b = (const float*)d_in[2];
    const float* conv2_w = (const float*)d_in[3];
    const float* conv2_b = (const float*)d_in[4];
    const float* conv3_w = (const float*)d_in[5];
    const float* conv3_b = (const float*)d_in[6];
    const float* lin1_w  = (const float*)d_in[7];
    const float* lin1_b  = (const float*)d_in[8];
    const float* heads_w = (const float*)d_in[9];
    const float* heads_b = (const float*)d_in[10];

    // workspace layout (f16 elements), ~49 MB total
    _Float16* ws = (_Float16*)d_ws;
    size_t o = 0;
    _Float16* w1h  = ws + o; o += (size_t)32 * 256;
    _Float16* w2h  = ws + o; o += (size_t)64 * 512;
    _Float16* w3h  = ws + o; o += (size_t)64 * 576;
    _Float16* wLh  = ws + o; o += (size_t)512 * 3136;
    _Float16* wHT  = ws + o; o += (size_t)200 * 18 * 512;  // transposed heads_w
    _Float16* act1 = ws + o; o += (size_t)1024 * 32 * 400;
    _Float16* act2 = ws + o; o += (size_t)1024 * 64 * 81;
    _Float16* act3 = ws + o; o += (size_t)1024 * 3136;
    _Float16* hbuf = ws + o; o += (size_t)1024 * 512;

    cvt_f32_to_f16<<<(8192 + 255) / 256, 256, 0, stream>>>(conv1_w, w1h, 8192);
    cvt_f32_to_f16<<<(32768 + 255) / 256, 256, 0, stream>>>(conv2_w, w2h, 32768);
    cvt_f32_to_f16<<<(36864 + 255) / 256, 256, 0, stream>>>(conv3_w, w3h, 36864);
    cvt_f32_to_f16<<<(1605632 + 255) / 256, 256, 0, stream>>>(lin1_w, wLh, 1605632);
    cvt_heads_T<<<(1843200 + 255) / 256, 256, 0, stream>>>(heads_w, wHT);

    // conv1: M=409600, K=256, N=32
    conv_gemm_wmma<32, float, false><<<dim3(6400, 1), 128, 0, stream>>>(
        x, w1h, conv1_b, act1,
        256, 32, 400, 20, 4, 64, 8, 4 * 84 * 84, 84 * 84, 84);

    // conv2: M=82944, K=512, N=64
    conv_gemm_wmma<64, _Float16, false><<<dim3(1296, 1), 128, 0, stream>>>(
        act1, w2h, conv2_b, act2,
        512, 64, 81, 9, 2, 16, 4, 32 * 400, 400, 20);

    // conv3: M=50176, K=576, N=64
    conv_gemm_wmma<64, _Float16, false><<<dim3(784, 1), 128, 0, stream>>>(
        act2, w3h, conv3_b, act3,
        576, 64, 49, 7, 1, 9, 3, 64 * 81, 81, 9);

    // lin1: M=1024, K=3136, N=512 (vector A staging)
    conv_gemm_wmma<64, _Float16, true><<<dim3(16, 8), 128, 0, stream>>>(
        act3, wLh, lin1_b, hbuf,
        3136, 512, 1, 1, 1, 1, 1, 3136, 1, 1);

    heads_wmma<<<dim3(16, 200), 128, 0, stream>>>(hbuf, wHT, heads_b, (float*)d_out);

    (void)in_sizes; (void)n_in; (void)out_size; (void)ws_size;
}
